// SkeletonRNN_85117661872789
// MI455X (gfx1250) — compile-verified, matrix operand order checked by reading, not run on hardware
//
#include <hip/hip_runtime.h>
#include <hip/hip_bf16.h>

// ---------------------------------------------------------------------------
// SkeletonRNN for MI455X (gfx1250): per-step WMMA-bf16 gates GEMM (TDM-staged
// A slab, 4 N-tiles per wave, fully unrolled K loop) + LSTM cell + LDS-resident
// conv decoder. Recurrence => 3 kernels per timestep on stream.
// ---------------------------------------------------------------------------

typedef __attribute__((ext_vector_type(16))) __bf16 v16bf;
typedef __attribute__((ext_vector_type(8)))  float  v8f;
typedef unsigned int u32x4 __attribute__((ext_vector_type(4)));
typedef int          i32x8 __attribute__((ext_vector_type(8)));
typedef int          i32x4 __attribute__((ext_vector_type(4)));

#define BS_   16
#define SEQ_  128
#define R_    18
#define D_    3
#define HS_   1024
#define NG_   4096      // 4*HS
#define MR_   (BS_*R_)  // 288 rows

__device__ __forceinline__ unsigned short f2bf(float f) {
    union { float f; unsigned int u; } v; v.f = f;
    unsigned int r = v.u + 0x7FFFu + ((v.u >> 16) & 1u);   // round-to-nearest-even
    return (unsigned short)(r >> 16);
}

// raw LDS byte offset of a __shared__ object (ptrtoint in addrspace(3))
__device__ __forceinline__ unsigned lds_offset_of(const void* p) {
    return (unsigned)(size_t)(const __attribute__((address_space(3))) char*)p;
}

// --- one-time conversions / init --------------------------------------------

__global__ void convert_Ut(const float* __restrict__ U, unsigned short* __restrict__ Ut) {
    int idx = blockIdx.x * 256 + threadIdx.x;          // idx = k*4096 + n (coalesced read)
    int k = idx >> 12, n = idx & 4095;
    Ut[(size_t)n * HS_ + k] = f2bf(U[idx]);            // Ut is N-major: [n][k]
}

__global__ void init_state(unsigned short* __restrict__ hbf, float* __restrict__ cst) {
    int idx = blockIdx.x * 256 + threadIdx.x;          // < MR_*HS_
    hbf[idx] = 0;
    cst[idx] = 0.f;
}

// --- gates = h_bf16 @ U_bf16 (WMMA) + bias + x_ref @ W ----------------------
// grid: (16, 18), block: 128 (4 waves). Block covers 16 rows x 256 cols; each
// wave owns 4 adjacent 16x16 N-tiles sharing one A fragment per K-chunk.
// The 16x1024 bf16 A slab is loaded to LDS once by the Tensor Data Mover.

__global__ void __launch_bounds__(128)
gates_gemm(const unsigned short* __restrict__ hbf,   // 288 x 1024 bf16 (row-major)
           const unsigned short* __restrict__ Ut,    // 4096 x 1024 bf16 (N-major)
           const float* __restrict__ Wm,             // 3 x 4096
           const float* __restrict__ bias,           // 4096
           const float* __restrict__ x,              // (BS,SEQ,R,3)
           const float* __restrict__ preds,          // d_out preds base (BS,SEQ,R,3)
           float* __restrict__ gates,                // 288 x 4096
           int t) {
    const int lane = threadIdx.x & 31;
    const int wave = threadIdx.x >> 5;
    const int m0 = blockIdx.y * 16;
    const int n0 = blockIdx.x * 256 + wave * 64;     // first of 4 N-tiles

    // whole A slab for this block: 16 rows x 1024 K of bf16 = 32 KB LDS
    __shared__ __align__(16) unsigned short Asl[16 * HS_];

#if __has_builtin(__builtin_amdgcn_tensor_load_to_lds)
    if (threadIdx.x < 32) {
        // Build D# per CDNA5 ISA ch.8: 2D tile 1024x16 of 2-byte elements.
        unsigned long long ga = (unsigned long long)(size_t)(hbf + (size_t)m0 * HS_);
        unsigned ldsb = lds_offset_of(Asl);
        u32x4 g0;
        g0[0] = 1u;                                   // count=1 (valid), user mode
        g0[1] = ldsb;                                 // lds_addr
        g0[2] = (unsigned)(ga & 0xFFFFFFFFu);         // global_addr[31:0]
        g0[3] = (unsigned)((ga >> 32) & 0x01FFFFFFu)  // global_addr[56:32]
                | (2u << 30);                         // type=2 ("image")
        i32x8 g1;
        g1[0] = (int)(1u << 16);                      // wg_mask=0, data_size=1 (2B)
        g1[1] = (int)(HS_ << 16);                     // tensor_dim0[15:0]=1024
        g1[2] = (int)(16u << 16);                     // tensor_dim0 msbs=0, tensor_dim1=16
        g1[3] = (int)((unsigned)HS_ << 16);           // tensor_dim1 msbs=0, tile_dim0=1024
        g1[4] = 16;                                   // tile_dim1=16, tile_dim2=0
        g1[5] = HS_;                                  // tensor_dim0_stride[31:0]=1024
        g1[6] = 0;                                    // stride msbs / dim1_stride lsbs
        g1[7] = 0;
        i32x4 gz4 = {0, 0, 0, 0};
        i32x8 gz8 = {0, 0, 0, 0, 0, 0, 0, 0};
        // 6-arg form (clang-23 / therock-10.0 HIP headers)
        __builtin_amdgcn_tensor_load_to_lds(g0, g1, gz4, gz4, gz8, 0);
#if __has_builtin(__builtin_amdgcn_s_wait_tensorcnt)
        __builtin_amdgcn_s_wait_tensorcnt(0);
#else
        asm volatile("s_wait_tensorcnt 0x0" ::: "memory");
#endif
    }
#else
    // fallback: cooperative copy (8B per thread per iteration)
    for (int i = threadIdx.x; i < 16 * HS_ / 4; i += 128)
        ((uint2*)Asl)[i] = ((const uint2*)(hbf + (size_t)m0 * HS_))[i];
#endif
    __syncthreads();

    const int arow = lane & 15;            // M within tile
    const int hi   = lane >> 4;            // half-wave selector
    const int ka   = hi * 8;               // K sub-offset per ISA 16-bit A/B layout
    const int nlan = n0 + (lane & 15);     // B/C column of tile 0 for this lane

    v8f acc[4] = {{}, {}, {}, {}};
    union AB { v16bf v; unsigned int u[8]; };

    // B stream base pointers (one per N-tile), K advances by constant offsets
    const unsigned short* Bt0 = &Ut[(size_t)(nlan)      * HS_ + ka];
    const unsigned short* Bt1 = &Ut[(size_t)(nlan + 16) * HS_ + ka];
    const unsigned short* Bt2 = &Ut[(size_t)(nlan + 32) * HS_ + ka];
    const unsigned short* Bt3 = &Ut[(size_t)(nlan + 48) * HS_ + ka];
    const unsigned short* Bp[4] = {Bt0, Bt1, Bt2, Bt3};

#pragma unroll
    for (int kc = 0; kc < HS_ / 32; ++kc) {
        const int k0 = kc * 32;
        AB a;
#pragma unroll
        for (int vv = 0; vv < 4; ++vv) {
            a.u[vv]     = *(const unsigned int*)&Asl[arow * HS_ + k0 + ka + 2 * vv];
            a.u[4 + vv] = *(const unsigned int*)&Asl[arow * HS_ + k0 + ka + 16 + 2 * vv];
        }
#pragma unroll
        for (int tt = 0; tt < 4; ++tt) {
            const unsigned short* Bt = Bp[tt] + k0;
            AB b;
#pragma unroll
            for (int vv = 0; vv < 4; ++vv) {
                b.u[vv]     = *(const unsigned int*)&Bt[2 * vv];
                b.u[4 + vv] = *(const unsigned int*)&Bt[16 + 2 * vv];
            }
            if (kc + 1 < HS_ / 32)
                __builtin_prefetch(&Bt[32], 0, 1);           // global_prefetch_b8
            acc[tt] = __builtin_amdgcn_wmma_f32_16x16x32_bf16(
                          false, a.v, false, b.v, (short)0, acc[tt], false, false);
        }
    }

    // Epilogue: + bias + x_ref @ W  (x_ref = refine_input of reference)
    float bn[4], w0n[4], w1n[4], w2n[4];
#pragma unroll
    for (int tt = 0; tt < 4; ++tt) {
        int n = nlan + 16 * tt;
        bn[tt]  = bias[n];
        w0n[tt] = Wm[0 * NG_ + n];
        w1n[tt] = Wm[1 * NG_ + n];
        w2n[tt] = Wm[2 * NG_ + n];
    }
#pragma unroll
    for (int j = 0; j < 8; ++j) {
        int m = m0 + j + hi * 8;                    // C/D layout: vgpr j, half-waves
        int bb = m / R_, rr = m % R_;
        const float* xp = &x[(((size_t)bb * SEQ_ + t) * R_ + rr) * D_];
        float x0 = xp[0], x1 = xp[1], x2 = xp[2];
        if (t > 0 && x0 == -1.f && x1 == -1.f && x2 == -1.f) {
            const float* pp = &preds[(((size_t)bb * SEQ_ + (t - 1)) * R_ + rr) * D_];
            x0 = pp[0]; x1 = pp[1]; x2 = pp[2];
        }
#pragma unroll
        for (int tt = 0; tt < 4; ++tt) {
            int n = nlan + 16 * tt;
            gates[(size_t)m * NG_ + n] =
                acc[tt][j] + bn[tt] + x0 * w0n[tt] + x1 * w1n[tt] + x2 * w2n[tt];
        }
    }
}

// --- LSTM cell elementwise ---------------------------------------------------

__global__ void lstm_update(const float* __restrict__ gates,
                            float* __restrict__ cst,
                            unsigned short* __restrict__ hbf,
                            float* __restrict__ hidden,   // d_out hidden base
                            int t) {
    int idx = blockIdx.x * 256 + threadIdx.x;     // < MR_*HS_
    int m = idx >> 10, k = idx & (HS_ - 1);
    const float* g = &gates[(size_t)m * NG_];
    float i_t = 1.f / (1.f + __expf(-g[k]));
    float f_t = 1.f / (1.f + __expf(-g[HS_ + k]));
    float g_t = tanhf(g[2 * HS_ + k]);
    float o_t = 1.f / (1.f + __expf(-g[3 * HS_ + k]));
    float c = f_t * cst[idx] + i_t * g_t;
    float h = o_t * tanhf(c);
    cst[idx] = c;
    hbf[idx] = f2bf(h);
    int bb = m / R_, rr = m % R_;
    hidden[((((size_t)bb * SEQ_ + t) * R_ + rr) * HS_) + k] = h;
}

// --- conv/pool decoder: one workgroup per (batch, joint) row -----------------
// Chain (width): 1024 -c3p1-relu-pool2-> 512 -c3p1-relu-pool2-> 256
//   -c5p2-relu-pool8-> 32 -c3p1-relu-pool3-> 10 -c3p1-relu-pool2-> 5
//   -c3p1s2-> 3.   Channels: 1->16->32->64->32->16->1.

__global__ void __launch_bounds__(256)
decode(const float* __restrict__ hidden,
       const float* __restrict__ w1, const float* __restrict__ b1,
       const float* __restrict__ w2, const float* __restrict__ b2,
       const float* __restrict__ w3, const float* __restrict__ b3,
       const float* __restrict__ w4, const float* __restrict__ b4,
       const float* __restrict__ w5, const float* __restrict__ b5,
       const float* __restrict__ w6, const float* __restrict__ b6,
       float* __restrict__ preds, int t) {
    __shared__ float bufA[8192];   // ping
    __shared__ float bufB[8192];   // pong
    const int tid = threadIdx.x;
    const int bb = blockIdx.x / R_, rr = blockIdx.x % R_;

    // load h row into bufA[0..1023]
    const float* hrow = &hidden[(((size_t)bb * SEQ_ + t) * R_ + rr) * HS_];
    for (int i = tid; i < HS_; i += 256) bufA[i] = hrow[i];
    __syncthreads();

    // stage1: 1ch@1024 -> 16ch@512 (conv3 pad1 + relu + pool2), bufA -> bufB
    for (int idx = tid; idx < 16 * 512; idx += 256) {
        int c = idx >> 9, j = idx & 511;
        float k0 = w1[c * 3], k1 = w1[c * 3 + 1], k2 = w1[c * 3 + 2], bc = b1[c];
        int p0 = 2 * j;
        float im = (p0 - 1 >= 0) ? bufA[p0 - 1] : 0.f;
        float i0 = bufA[p0];
        float i1 = bufA[p0 + 1];
        float i2 = (p0 + 2 < 1024) ? bufA[p0 + 2] : 0.f;
        float a0 = bc + k0 * im + k1 * i0 + k2 * i1;
        float a1 = bc + k0 * i0 + k1 * i1 + k2 * i2;
        bufB[c * 512 + j] = fmaxf(0.f, fmaxf(a0, a1));
    }
    __syncthreads();

    // stage2: 16ch@512 -> 32ch@256, bufB -> bufA
    for (int idx = tid; idx < 32 * 256; idx += 256) {
        int c = idx >> 8, j = idx & 255;
        int p0 = 2 * j;
        float a0 = b2[c], a1 = b2[c];
        for (int ci = 0; ci < 16; ++ci) {
            const float* wp = &w2[(c * 16 + ci) * 3];
            const float* in = &bufB[ci * 512];
            float im = (p0 - 1 >= 0) ? in[p0 - 1] : 0.f;
            float i0 = in[p0];
            float i1 = in[p0 + 1];
            float i2 = (p0 + 2 < 512) ? in[p0 + 2] : 0.f;
            a0 += wp[0] * im + wp[1] * i0 + wp[2] * i1;
            a1 += wp[0] * i0 + wp[1] * i1 + wp[2] * i2;
        }
        bufA[c * 256 + j] = fmaxf(0.f, fmaxf(a0, a1));
    }
    __syncthreads();

    // stage3: 32ch@256 -> 64ch@32 (conv5 pad2 + relu + pool8), bufA -> bufB
    for (int idx = tid; idx < 64 * 32; idx += 256) {
        int c = idx >> 5, j = idx & 31;
        float mx = -1e30f;
        for (int q = 0; q < 8; ++q) {
            int p = 8 * j + q;
            float acc = b3[c];
            for (int ci = 0; ci < 32; ++ci) {
                const float* wp = &w3[(c * 32 + ci) * 5];
                const float* in = &bufA[ci * 256];
#pragma unroll
                for (int k = 0; k < 5; ++k) {
                    int ip = p + k - 2;
                    if (ip >= 0 && ip < 256) acc += wp[k] * in[ip];
                }
            }
            mx = fmaxf(mx, acc);
        }
        bufB[c * 32 + j] = fmaxf(0.f, mx);
    }
    __syncthreads();

    // stage4: 64ch@32 -> 32ch@10 (conv3 pad1 + relu + pool3), bufB -> bufA
    for (int idx = tid; idx < 32 * 10; idx += 256) {
        int c = idx / 10, j = idx % 10;
        float mx = -1e30f;
        for (int q = 0; q < 3; ++q) {
            int p = 3 * j + q;
            float acc = b4[c];
            for (int ci = 0; ci < 64; ++ci) {
                const float* wp = &w4[(c * 64 + ci) * 3];
                const float* in = &bufB[ci * 32];
#pragma unroll
                for (int k = 0; k < 3; ++k) {
                    int ip = p + k - 1;
                    if (ip >= 0 && ip < 32) acc += wp[k] * in[ip];
                }
            }
            mx = fmaxf(mx, acc);
        }
        bufA[c * 10 + j] = fmaxf(0.f, mx);
    }
    __syncthreads();

    // stage5: 32ch@10 -> 16ch@5 (conv3 pad1 + relu + pool2), bufA -> bufB
    for (int idx = tid; idx < 16 * 5; idx += 256) {
        int c = idx / 5, j = idx % 5;
        float mx = -1e30f;
        for (int q = 0; q < 2; ++q) {
            int p = 2 * j + q;
            float acc = b5[c];
            for (int ci = 0; ci < 32; ++ci) {
                const float* wp = &w5[(c * 32 + ci) * 3];
                const float* in = &bufA[ci * 10];
#pragma unroll
                for (int k = 0; k < 3; ++k) {
                    int ip = p + k - 1;
                    if (ip >= 0 && ip < 10) acc += wp[k] * in[ip];
                }
            }
            mx = fmaxf(mx, acc);
        }
        bufB[c * 5 + j] = fmaxf(0.f, mx);
    }
    __syncthreads();

    // stage6: 16ch@5 -> 1ch@3 (conv3 pad1 stride2, NO relu), bufB -> out
    if (tid < 3) {
        int j = tid;
        float acc = b6[0];
        for (int ci = 0; ci < 16; ++ci) {
            const float* wp = &w6[ci * 3];
            const float* in = &bufB[ci * 5];
#pragma unroll
            for (int k = 0; k < 3; ++k) {
                int ip = 2 * j + k - 1;
                if (ip >= 0 && ip < 5) acc += wp[k] * in[ip];
            }
        }
        preds[(((size_t)bb * SEQ_ + t) * R_ + rr) * D_ + j] = acc;
    }
}

// ---------------------------------------------------------------------------

extern "C" void kernel_launch(void* const* d_in, const int* in_sizes, int n_in,
                              void* d_out, int out_size, void* d_ws, size_t ws_size,
                              hipStream_t stream) {
    const float* x    = (const float*)d_in[0];
    const float* W    = (const float*)d_in[1];
    const float* U    = (const float*)d_in[2];
    const float* bias = (const float*)d_in[3];
    const float* w1 = (const float*)d_in[4],  *b1 = (const float*)d_in[5];
    const float* w2 = (const float*)d_in[6],  *b2 = (const float*)d_in[7];
    const float* w3 = (const float*)d_in[8],  *b3 = (const float*)d_in[9];
    const float* w4 = (const float*)d_in[10], *b4 = (const float*)d_in[11];
    const float* w5 = (const float*)d_in[12], *b5 = (const float*)d_in[13];
    const float* w6 = (const float*)d_in[14], *b6 = (const float*)d_in[15];

    float* preds  = (float*)d_out;                        // (16,128,18,3)
    float* hidden = preds + (size_t)BS_ * SEQ_ * R_ * D_; // (16,128,18,1024)

    // workspace carve-up
    char* ws = (char*)d_ws;
    unsigned short* Ut  = (unsigned short*)ws;                              // 8 MB
    unsigned short* hbf = (unsigned short*)(ws + (size_t)NG_ * HS_ * 2);    // 576 KB
    float* cst   = (float*)(ws + (size_t)NG_ * HS_ * 2 + (size_t)MR_ * HS_ * 2);
    float* gates = cst + (size_t)MR_ * HS_;                                 // 4.5 MB

    convert_Ut<<<(NG_ * HS_) / 256, 256, 0, stream>>>(U, Ut);
    init_state<<<(MR_ * HS_) / 256, 256, 0, stream>>>(hbf, cst);

    for (int t = 0; t < SEQ_; ++t) {
        gates_gemm<<<dim3(16, 18), 128, 0, stream>>>(hbf, Ut, W, bias, x, preds,
                                                     gates, t);
        lstm_update<<<(MR_ * HS_) / 256, 256, 0, stream>>>(gates, cst, hbf,
                                                           hidden, t);
        decode<<<MR_, 256, 0, stream>>>(hidden, w1, b1, w2, b2, w3, b3,
                                        w4, b4, w5, b5, w6, b6, preds, t);
    }
}
